// RelevancePropagationTAGCN_38551626449046
// MI455X (gfx1250) — matrix operand order repaired and checked by reading, not compiled, not run on hardware
//
#include <hip/hip_runtime.h>
#include <hip/hip_bf16.h>

// ---------------------------------------------------------------------------
// LRP-TAGConv for MI455X (gfx1250, wave32, WMMA).
//
// Pipeline (all on `stream`, workspace-resident):
//   k_dinv        : deg = colsum(adj); dinv = rsqrt(max(deg,1))
//   k_buildB      : B = A^T in bf16 via LDS-tiled transpose (B[x][i]=dinv_x*dinv_i*adj[i][x])
//   k_wprep       : Wt[c][j] = |W[j][c]| (bf16) + wsum[k][j]
//   k_colmv x3    : rowsum chain r1=A·1, r2=A·r1, r3=A·r2 (coalesced column sums of B)
//   k_s           : s = r / denom (bf16)
//   gemm_nt_wmma  : tT (512x4096) = Wt (512x128) · s^T       [WMMA bf16]
//   gemm_nt_wmma  : Y2T = t3T·A + t2T                        [WMMA bf16]
//   gemm_nt_wmma  : Y1T = Y2T·A + t1T                        [WMMA bf16]
//   gemm_nt_wmma  : finalT = Y1T·A + t0T (fp32 out)          [WMMA bf16]
//   k_transpose   : out = finalT^T
// ---------------------------------------------------------------------------

typedef __attribute__((ext_vector_type(16))) __bf16 v16bf;
typedef __attribute__((ext_vector_type(8)))  __bf16 v8bf;
typedef __attribute__((ext_vector_type(8)))  float  v8f;

#define NNODES 4096
#define DFEAT  128
#define WCOLS  512   // (K+1)*DFEAT

// ---------------------------------------------------------------------------
// NT GEMM with WMMA:  C[m,n] = sum_k Amat[m,k] * Bmat[n,k]  (+ addT[m,n])
// Amat: M x K row-major bf16;  Bmat: N x K row-major bf16 (consumed transposed)
// Per wave: 16(M) x 64(N) tile of C, K-loop step 32.
// Operand register layouts (A and B share a contiguous K order per lane half):
//   A-frag  lane l: row m0+(l&15); v0-7 pack K {k0+16*(l>>4) .. +15}
//                                                     -> one contiguous 32B load
//   B-frag  lane l: col n0+(l&15); v0-7 pack K {k0+16*(l>>4) .. +15}
//                                                     -> one contiguous 32B load
//   C/D     lane l, vgpr g: row m0+g+8*(l>>4), col n0+(l&15)
// ---------------------------------------------------------------------------
__global__ __launch_bounds__(128)
void gemm_nt_wmma(const __bf16* __restrict__ Amat,
                  const __bf16* __restrict__ Bmat,
                  const __bf16* __restrict__ addT,   // M x N bf16 addend, or null
                  __bf16*       __restrict__ outB,   // M x N bf16 result, or null
                  float*        __restrict__ outF,   // M x N f32 result, or null
                  int M, int N, int K)
{
  const int lane   = threadIdx.x & 31;
  const int wave   = blockIdx.x * (blockDim.x >> 5) + (threadIdx.x >> 5);
  const int mTiles = M >> 4;
  const int m0     = (wave % mTiles) << 4;
  const int n0     = (wave / mTiles) << 6;           // 64-wide N group
  if (n0 >= N) return;

  const int lhalf = lane >> 4;                        // 0 | 1
  const int lmod  = lane & 15;

  v8f acc[4];
  #pragma unroll
  for (int t = 0; t < 4; ++t) {
    if (addT != nullptr) {
      const int col = n0 + t * 16 + lmod;
      #pragma unroll
      for (int g = 0; g < 8; ++g)
        acc[t][g] = (float)addT[(size_t)(m0 + g + (lhalf << 3)) * N + col];
    } else {
      #pragma unroll
      for (int g = 0; g < 8; ++g) acc[t][g] = 0.0f;
    }
  }

  const __bf16* aRow = Amat + (size_t)(m0 + lmod) * K;

  for (int k0 = 0; k0 < K; k0 += 32) {
    // A fragment: one 32B contiguous load per lane
    const v16bf afrag = *(const v16bf*)(aRow + k0 + (lhalf << 4));

    #pragma unroll
    for (int t = 0; t < 4; ++t) {
      // B fragment: one 32B contiguous load per lane
      const __bf16* bRow = Bmat + (size_t)(n0 + t * 16 + lmod) * K;
      const v16bf bfrag  = *(const v16bf*)(bRow + k0 + (lhalf << 4));
      acc[t] = __builtin_amdgcn_wmma_f32_16x16x32_bf16(
                   false, afrag, false, bfrag, (short)0, acc[t], false, false);
    }
  }

  #pragma unroll
  for (int t = 0; t < 4; ++t) {
    const int col = n0 + t * 16 + lmod;
    #pragma unroll
    for (int g = 0; g < 8; ++g) {
      const size_t idx = (size_t)(m0 + g + (lhalf << 3)) * N + col;
      if (outF != nullptr) outF[idx] = acc[t][g];
      else                 outB[idx] = (__bf16)acc[t][g];
    }
  }
}

// deg[x] = sum_i adj[i][x]; dinv = rsqrt(max(deg,1)). Coalesced column sums.
__global__ void k_dinv(const int* __restrict__ adj, float* __restrict__ dinv, int n)
{
  const int x = blockIdx.x * blockDim.x + threadIdx.x;
  int deg = 0;
  for (int i = 0; i < n; ++i) deg += adj[(size_t)i * n + x];
  float d = (float)deg;
  if (d < 1.0f) d = 1.0f;
  dinv[x] = rsqrtf(d);
}

// B = A^T (bf16) via 32x32 LDS-tiled transpose. B[x][i] = dinv_x * dinv_i * adj[i][x]
__global__ void k_buildB(const int* __restrict__ adj, const float* __restrict__ dinv,
                         __bf16* __restrict__ B, int n)
{
  __shared__ float tile[32][33];
  const int i0 = blockIdx.x * 32;       // adj row block
  const int x0 = blockIdx.y * 32;       // adj col block
  for (int r = threadIdx.y; r < 32; r += blockDim.y) {
    const int i = i0 + r, x = x0 + threadIdx.x;
    tile[r][threadIdx.x] = (float)adj[(size_t)i * n + x] * dinv[i] * dinv[x];
  }
  __syncthreads();
  for (int r = threadIdx.y; r < 32; r += blockDim.y) {
    const int x = x0 + r, i = i0 + threadIdx.x;
    B[(size_t)x * n + i] = (__bf16)tile[threadIdx.x][r];
  }
}

// Wt[c][j] = |W[j][c]| (bf16, 512x128) and wsum[k][j] = sum_y |W[j][k*128+y]|
__global__ void k_wprep(const float* __restrict__ W, __bf16* __restrict__ Wt,
                        float* __restrict__ wsum)
{
  const int j = threadIdx.x;             // 0..127
  float sums[4] = {0.f, 0.f, 0.f, 0.f};
  for (int c = 0; c < WCOLS; ++c) {
    const float w = fabsf(W[j * WCOLS + c]);
    Wt[c * DFEAT + j] = (__bf16)w;       // coalesced across j
    sums[c >> 7] += w;
  }
  #pragma unroll
  for (int k = 0; k < 4; ++k) wsum[k * DFEAT + j] = sums[k];
}

// vout[i] = sum_x B[x][i] * (vin ? vin[x] : 1)   == (A @ v)[i] since B = A^T
__global__ void k_colmv(const __bf16* __restrict__ B, const float* __restrict__ vin,
                        float* __restrict__ vout, int n)
{
  const int i = blockIdx.x * blockDim.x + threadIdx.x;
  float s = 0.f;
  for (int x = 0; x < n; ++x) {
    const float b = (float)B[(size_t)x * n + i];
    s += (vin != nullptr) ? b * vin[x] : b;
  }
  vout[i] = s;
}

// s[i][j] = r[i][j] / (wsum0[j] + r1[i]*wsum1[j] + r2[i]*wsum2[j] + r3[i]*wsum3[j])
__global__ void k_s(const float* __restrict__ r, const float* __restrict__ r1,
                    const float* __restrict__ r2, const float* __restrict__ r3,
                    const float* __restrict__ wsum, __bf16* __restrict__ s)
{
  const int idx = blockIdx.x * blockDim.x + threadIdx.x;
  const int i = idx >> 7, j = idx & (DFEAT - 1);
  const float denom = wsum[j] + r1[i] * wsum[DFEAT + j]
                    + r2[i] * wsum[2 * DFEAT + j] + r3[i] * wsum[3 * DFEAT + j];
  s[idx] = (__bf16)(r[idx] / denom);
}

// out[x][y] = finalT[y][x]   (128 x 4096 -> 4096 x 128)
__global__ void k_transpose(const float* __restrict__ finalT, float* __restrict__ out, int n)
{
  __shared__ float tile[32][33];
  const int x0 = blockIdx.x * 32;        // finalT col block (node)
  const int y0 = blockIdx.y * 32;        // finalT row block (feature)
  for (int r = threadIdx.y; r < 32; r += blockDim.y)
    tile[r][threadIdx.x] = finalT[(size_t)(y0 + r) * n + x0 + threadIdx.x];
  __syncthreads();
  for (int r = threadIdx.y; r < 32; r += blockDim.y)
    out[(size_t)(x0 + r) * DFEAT + y0 + threadIdx.x] = tile[threadIdx.x][r];
}

static inline size_t alignUp(size_t v, size_t a) { return (v + a - 1) & ~(a - 1); }

extern "C" void kernel_launch(void* const* d_in, const int* in_sizes, int n_in,
                              void* d_out, int out_size, void* d_ws, size_t ws_size,
                              hipStream_t stream)
{
  (void)in_sizes; (void)n_in; (void)out_size; (void)ws_size;
  // inputs: a (unused values), r, adj, W
  const float* r   = (const float*)d_in[1];
  const int*   adj = (const int*)d_in[2];
  const float* W   = (const float*)d_in[3];
  float*       out = (float*)d_out;

  const int n = NNODES;
  char* ws = (char*)d_ws;
  size_t off = 0;
  auto carve = [&](size_t bytes) { void* p = ws + off; off += alignUp(bytes, 256); return p; };

  __bf16* B      = (__bf16*)carve((size_t)n * n * 2);            // 33.6 MB, A^T bf16
  float*  dinv   = (float*) carve((size_t)n * 4);
  float*  r1     = (float*) carve((size_t)n * 4);
  float*  r2     = (float*) carve((size_t)n * 4);
  float*  r3     = (float*) carve((size_t)n * 4);
  __bf16* Wt     = (__bf16*)carve((size_t)WCOLS * DFEAT * 2);
  float*  wsum   = (float*) carve((size_t)4 * DFEAT * 4);
  __bf16* s      = (__bf16*)carve((size_t)n * DFEAT * 2);
  __bf16* tT     = (__bf16*)carve((size_t)WCOLS * n * 2);        // 4 MB: [k*128+y][i]
  __bf16* Y2T    = (__bf16*)carve((size_t)DFEAT * n * 2);
  __bf16* Y1T    = (__bf16*)carve((size_t)DFEAT * n * 2);
  float*  finalT = (float*) carve((size_t)DFEAT * n * 4);

  // 1) degrees -> dinv
  k_dinv<<<n / 256, 256, 0, stream>>>(adj, dinv, n);

  // 2) B = A^T in bf16 (tiled transpose of adj with scaling)
  k_buildB<<<dim3(n / 32, n / 32), dim3(32, 8), 0, stream>>>(adj, dinv, B, n);

  // 3) |W|^T + per-hop weight sums
  k_wprep<<<1, DFEAT, 0, stream>>>(W, Wt, wsum);

  // 4) rowsum chain: r1 = A·1, r2 = A·r1, r3 = A·r2 (column sums of B, coalesced)
  k_colmv<<<n / 256, 256, 0, stream>>>(B, nullptr, r1, n);
  k_colmv<<<n / 256, 256, 0, stream>>>(B, r1, r2, n);
  k_colmv<<<n / 256, 256, 0, stream>>>(B, r2, r3, n);

  // 5) s = r / denom (bf16)
  k_s<<<(n * DFEAT) / 256, 256, 0, stream>>>(r, r1, r2, r3, wsum, s);

  // 6) tT (512 x 4096) = Wt (512x128) x s^T : waves = (512/16)*(4096/64) = 2048
  gemm_nt_wmma<<<2048 / 4, 128, 0, stream>>>(Wt, s, nullptr, tT, nullptr,
                                             WCOLS, n, DFEAT);

  const __bf16* t0T = tT;                      // hop-0 rows
  const __bf16* t1T = tT + (size_t)1 * DFEAT * n;
  const __bf16* t2T = tT + (size_t)2 * DFEAT * n;
  const __bf16* t3T = tT + (size_t)3 * DFEAT * n;

  // 7) Horner chain in transposed space: X^T·A == NT-GEMM(X^T, B)
  //    waves = (128/16)*(4096/64) = 512 -> 128 blocks of 4 waves
  gemm_nt_wmma<<<512 / 4, 128, 0, stream>>>(t3T, B, t2T, Y2T, nullptr, DFEAT, n, n);
  gemm_nt_wmma<<<512 / 4, 128, 0, stream>>>(Y2T, B, t1T, Y1T, nullptr, DFEAT, n, n);
  gemm_nt_wmma<<<512 / 4, 128, 0, stream>>>(Y1T, B, t0T, nullptr, finalT, DFEAT, n, n);

  // 8) out = finalT^T
  k_transpose<<<dim3(n / 32, DFEAT / 32), dim3(32, 8), 0, stream>>>(finalT, out, n);
}
